// ReserviorEncoder_24206435680634
// MI455X (gfx1250) — matrix operand order repaired and checked by reading, not compiled, not run on hardware
//
#include <hip/hip_runtime.h>
#include <math.h>

// ---------------------------------------------------------------------------
// Types / helpers
// ---------------------------------------------------------------------------
typedef __bf16 bf16;
typedef __attribute__((ext_vector_type(16))) __bf16 v16bf;
typedef __attribute__((ext_vector_type(8)))  float  v8f;

__device__ __forceinline__ bf16 f2bf(float f) {
    unsigned u = __float_as_uint(f);
    u = (u + 0x7FFFu + ((u >> 16) & 1u)) >> 16;   // round-to-nearest-even
    unsigned short s = (unsigned short)u;
    return __builtin_bit_cast(bf16, s);
}

// LDS tile strides (elements). 40*2B = 80B = 20 words: 20*n mod 64 is a
// permutation over n=0..15 -> conflict-free half-wave b128 reads.
#define ASTR 40
#define BSTR 40

// ---------------------------------------------------------------------------
// Elementwise prep kernels
// ---------------------------------------------------------------------------
__global__ void cvt_f32_bf16_kernel(const float* __restrict__ in,
                                    bf16* __restrict__ out, size_t n) {
    size_t i = (size_t)blockIdx.x * 256 + threadIdx.x;
    if (i < n) out[i] = f2bf(in[i]);
}

// x (64,20,4096) f32 -> xbf rows: [0..63]=(b,t=0), [64..1279]=((t-1)*64+b, t=1..19)
__global__ void prep_x_kernel(const float* __restrict__ x, bf16* __restrict__ xbf) {
    size_t j = (size_t)blockIdx.x * 256 + threadIdx.x;
    if (j >= (size_t)1280 * 4096) return;
    int i   = (int)(j & 4095);
    int row = (int)(j >> 12);
    int b, t;
    if (row < 64) { b = row; t = 0; }
    else          { int rr = row - 64; t = 1 + rr / 64; b = rr & 63; }
    xbf[j] = f2bf(x[((size_t)b * 20 + t) * 4096 + i]);
}

// H (b,t,8192) f32 -> h5d (b,c=8,t,32,32) bf16
__global__ void relayout_h_kernel(const float* __restrict__ Hst, bf16* __restrict__ h5d) {
    size_t i = (size_t)blockIdx.x * 256 + threadIdx.x;
    if (i >= (size_t)64 * 8 * 20 * 1024) return;
    int p = (int)(i & 1023);
    int t = (int)((i >> 10) % 20);
    int c = (int)((i / 20480) & 7);
    int b = (int)(i / 163840);
    h5d[i] = f2bf(Hst[(size_t)b * 163840 + (size_t)t * 8192 + c * 1024 + p]);
}

// conv weights (OC, Kreal) f32 -> (OC, Kpad) bf16, zero padded
__global__ void prep_wconv_kernel(const float* __restrict__ src, bf16* __restrict__ dst,
                                  int Kreal, int Kpad, int total) {
    int j = blockIdx.x * 256 + threadIdx.x;
    if (j >= total) return;
    int oc = j / Kpad, k = j % Kpad;
    dst[j] = (k < Kreal) ? f2bf(src[(size_t)oc * Kreal + k])
                         : __builtin_bit_cast(bf16, (unsigned short)0);
}

// lin_w (256,6272) f32 -> lwT (6272,256) bf16
__global__ void prep_linT_kernel(const float* __restrict__ lw, bf16* __restrict__ lwT) {
    int j = blockIdx.x * 256 + threadIdx.x;
    if (j >= 6272 * 256) return;
    int k = j >> 8, n = j & 255;
    lwT[j] = f2bf(lw[(size_t)n * 6272 + k]);
}

// ylin (1280,256) + lin_b -> split into mu / log_var halves of d_out
__global__ void split_out_kernel(const float* __restrict__ ylin,
                                 const float* __restrict__ lb,
                                 float* __restrict__ out) {
    int j = blockIdx.x * 256 + threadIdx.x;
    if (j >= 1280 * 256) return;
    int m = j >> 8, n = j & 255;
    float v = ylin[j] + lb[n];
    if (n < 128) out[(size_t)m * 128 + n]          = v;
    else         out[163840 + (size_t)m * 128 + (n - 128)] = v;
}

// ---------------------------------------------------------------------------
// Tiled bf16 GEMM with f32 accumulation.
//   C(MxN) = A(MxK) @ B(KxN)  [+ Dsrc]  [tanh]  -> C f32 (ldc) and/or Cbf bf16
// Block: 256 thr = 8 waves; tile 64x128; wave tile 32x32 (2x2 WMMA frags).
// LDS: A row-major (k minor), B transposed (n major, k minor) so both A and B
// fragments are 2x ds_load_b128 per 16x16 tile.
// M%64==0, N%128==0, K%32==0 required (true for all shapes used here).
// ---------------------------------------------------------------------------
__global__ __launch_bounds__(256)
void gemm_bf16_wmma(const bf16* __restrict__ A, int lda,
                    const bf16* __restrict__ Bm, int ldb,
                    float* __restrict__ C, int ldc,
                    const float* __restrict__ Dsrc, int ldd,
                    bf16* __restrict__ Cbf, int ldcbf,
                    int K, int do_tanh) {
    __shared__ __align__(16) bf16 As[64 * ASTR];
    __shared__ __align__(16) bf16 Bs[128 * BSTR];   // transposed: [n][k]

    const int tid  = threadIdx.x;
    const int lane = tid & 31;
    const int wave = tid >> 5;
    const int wy = wave >> 2, wx = wave & 3;
    const int mbase = blockIdx.y * 64;
    const int nbase = blockIdx.x * 128;
    const int halfsel = lane >> 4;
    const int lrow    = lane & 15;

    v8f acc[2][2] = {};

    const int arow = tid >> 2, acol = (tid & 3) << 3;   // A: 8 bf16 / thread
    const int brow = tid >> 3, bcol = (tid & 7) << 4;   // B: 16 bf16 / thread

    union Pack { uint4 u; bf16 h[8]; };

    for (int k0 = 0; k0 < K; k0 += 32) {
        __syncthreads();
        *(uint4*)(As + arow * ASTR + acol) =
            *(const uint4*)(A + (size_t)(mbase + arow) * lda + (k0 + acol));
        {
            const bf16* bsrc = Bm + (size_t)(k0 + brow) * ldb + (nbase + bcol);
            Pack p0, p1;
            p0.u = *(const uint4*)bsrc;
            p1.u = *(const uint4*)(bsrc + 8);
            #pragma unroll
            for (int i = 0; i < 8; ++i) {
                Bs[(bcol + i)     * BSTR + brow] = p0.h[i];
                Bs[(bcol + 8 + i) * BSTR + brow] = p1.h[i];
            }
        }
        __syncthreads();

        #pragma unroll
        for (int im = 0; im < 2; ++im) {
            v16bf afrag;
            const int mrow = wy * 32 + im * 16 + lrow;
            #pragma unroll
            for (int e = 0; e < 16; ++e) {
                const int kk = (halfsel << 4) + e;
                afrag[e] = As[mrow * ASTR + kk];
            }
            #pragma unroll
            for (int jn = 0; jn < 2; ++jn) {
                v16bf bfrag;
                const int ncol = wx * 32 + jn * 16 + lrow;
                #pragma unroll
                for (int e = 0; e < 16; ++e)
                    bfrag[e] = Bs[ncol * BSTR + (halfsel << 4) + e];
                acc[im][jn] = __builtin_amdgcn_wmma_f32_16x16x32_bf16(
                    false, afrag, false, bfrag, (short)0, acc[im][jn], false, false);
            }
        }
    }

    // Epilogue: materialize, then uniformly-guarded passes.
    float vals[2][2][8];
    #pragma unroll
    for (int im = 0; im < 2; ++im)
        #pragma unroll
        for (int jn = 0; jn < 2; ++jn)
            #pragma unroll
            for (int r = 0; r < 8; ++r)
                vals[im][jn][r] = acc[im][jn][r];

    const int m0 = mbase + wy * 32 + (halfsel << 3);
    const int n0 = nbase + wx * 32 + lrow;
    if (Dsrc) {
        #pragma unroll
        for (int im = 0; im < 2; ++im)
            #pragma unroll
            for (int jn = 0; jn < 2; ++jn)
                #pragma unroll
                for (int r = 0; r < 8; ++r)
                    vals[im][jn][r] +=
                        Dsrc[(size_t)(m0 + im * 16 + r) * ldd + (n0 + jn * 16)];
    }
    if (do_tanh) {
        #pragma unroll
        for (int im = 0; im < 2; ++im)
            #pragma unroll
            for (int jn = 0; jn < 2; ++jn)
                #pragma unroll
                for (int r = 0; r < 8; ++r)
                    vals[im][jn][r] = tanhf(vals[im][jn][r]);
    }
    if (C) {
        #pragma unroll
        for (int im = 0; im < 2; ++im)
            #pragma unroll
            for (int jn = 0; jn < 2; ++jn)
                #pragma unroll
                for (int r = 0; r < 8; ++r)
                    C[(size_t)(m0 + im * 16 + r) * ldc + (n0 + jn * 16)] =
                        vals[im][jn][r];
    }
    if (Cbf) {
        #pragma unroll
        for (int im = 0; im < 2; ++im)
            #pragma unroll
            for (int jn = 0; jn < 2; ++jn)
                #pragma unroll
                for (int r = 0; r < 8; ++r)
                    Cbf[(size_t)(m0 + im * 16 + r) * ldcbf + (n0 + jn * 16)] =
                        f2bf(vals[im][jn][r]);
    }
}

// ---------------------------------------------------------------------------
// Implicit-GEMM 3D conv (stride D=1, H=W=2) with WMMA.
//   In (B,IC,D,H,W) bf16, Wm (OC,Kpad) bf16 (zero-padded), Out (B,OC,D,OH,OW) bf16
//   M=OC (tile 64), N=B*D*OH*OW (tile 128), K over (ic,kd,kh,kw) padded to 32.
// ---------------------------------------------------------------------------
__global__ __launch_bounds__(256)
void conv_igemm_wmma(const bf16* __restrict__ In, const bf16* __restrict__ Wm,
                     const float* __restrict__ bias, bf16* __restrict__ Out,
                     int IC, int D, int H, int W,
                     int OC, int OH, int OW,
                     int KD, int KH, int KW,
                     int padD, int padH, int padW,
                     int Kreal, int Kpad) {
    __shared__ __align__(16) bf16 As[64 * ASTR];
    __shared__ __align__(16) bf16 Bs[128 * BSTR];   // transposed: [n][k]

    const int tid  = threadIdx.x;
    const int lane = tid & 31;
    const int wave = tid >> 5;
    const int wy = wave >> 2, wx = wave & 3;
    const int mbase = blockIdx.y * 64;
    const int nbase = blockIdx.x * 128;
    const int halfsel = lane >> 4;
    const int lrow    = lane & 15;

    v8f acc[2][2] = {};

    const int arow = tid >> 2, acol = (tid & 3) << 3;
    const int brow = tid >> 3, bcol0 = (tid & 7) << 4;

    // Hoist im2col column decomposition (n fixed per thread across K loop)
    int inBase[16], idb[16], ihb[16], iwb[16];
    #pragma unroll
    for (int i = 0; i < 16; ++i) {
        int n  = nbase + bcol0 + i;
        int ow = n % OW;
        int t1 = n / OW;
        int oh = t1 % OH;
        int t2 = t1 / OH;
        int d  = t2 % D;
        int b  = t2 / D;
        inBase[i] = ((b * IC) * D) * H * W;
        idb[i] = d - padD;
        ihb[i] = oh * 2 - padH;
        iwb[i] = ow * 2 - padW;
    }

    for (int k0 = 0; k0 < Kpad; k0 += 32) {
        __syncthreads();
        *(uint4*)(As + arow * ASTR + acol) =
            *(const uint4*)(Wm + (size_t)(mbase + arow) * Kpad + (k0 + acol));
        {
            int k = k0 + brow;
            int ic = 0, kd = 0, kh = 0, kw = 0;
            bool kvalid = (k < Kreal);
            if (kvalid) {
                ic = k / (KD * KH * KW);
                int r  = k % (KD * KH * KW);
                kd = r / (KH * KW);
                int r2 = r % (KH * KW);
                kh = r2 / KW;
                kw = r2 % KW;
            }
            #pragma unroll
            for (int i = 0; i < 16; ++i) {
                bf16 v = __builtin_bit_cast(bf16, (unsigned short)0);
                if (kvalid) {
                    int id = idb[i] + kd, ih = ihb[i] + kh, iw = iwb[i] + kw;
                    if ((unsigned)id < (unsigned)D && (unsigned)ih < (unsigned)H &&
                        (unsigned)iw < (unsigned)W)
                        v = In[(size_t)inBase[i] + ((size_t)ic * D + id) * H * W
                               + (size_t)ih * W + iw];
                }
                Bs[(bcol0 + i) * BSTR + brow] = v;
            }
        }
        __syncthreads();

        #pragma unroll
        for (int im = 0; im < 2; ++im) {
            v16bf afrag;
            const int mrow = wy * 32 + im * 16 + lrow;
            #pragma unroll
            for (int e = 0; e < 16; ++e) {
                const int kk = (halfsel << 4) + e;
                afrag[e] = As[mrow * ASTR + kk];
            }
            #pragma unroll
            for (int jn = 0; jn < 2; ++jn) {
                v16bf bfrag;
                const int ncol = wx * 32 + jn * 16 + lrow;
                #pragma unroll
                for (int e = 0; e < 16; ++e)
                    bfrag[e] = Bs[ncol * BSTR + (halfsel << 4) + e];
                acc[im][jn] = __builtin_amdgcn_wmma_f32_16x16x32_bf16(
                    false, afrag, false, bfrag, (short)0, acc[im][jn], false, false);
            }
        }
    }

    #pragma unroll
    for (int im = 0; im < 2; ++im)
        #pragma unroll
        for (int jn = 0; jn < 2; ++jn)
            #pragma unroll
            for (int r = 0; r < 8; ++r) {
                const int oc = mbase + wy * 32 + im * 16 + r + (halfsel << 3);
                const int n  = nbase + wx * 32 + jn * 16 + lrow;
                float v = acc[im][jn][r] + bias[oc];
                v = v > 0.f ? v : 0.01f * v;          // leaky_relu(0.01)
                int ow = n % OW;
                int t1 = n / OW;
                int oh = t1 % OH;
                int t2 = t1 / OH;
                int d  = t2 % D;
                int b  = t2 / D;
                Out[((((size_t)b * OC + oc) * D + d) * OH + oh) * OW + ow] = f2bf(v);
            }
}

// ---------------------------------------------------------------------------
// Host launcher
// ---------------------------------------------------------------------------
extern "C" void kernel_launch(void* const* d_in, const int* in_sizes, int n_in,
                              void* d_out, int out_size, void* d_ws, size_t ws_size,
                              hipStream_t stream) {
    const float* x   = (const float*)d_in[0];
    const float* Win = (const float*)d_in[1];
    const float* U   = (const float*)d_in[2];
    const float* Wr  = (const float*)d_in[3];
    const float* c1w = (const float*)d_in[4];
    const float* c1b = (const float*)d_in[5];
    const float* c2w = (const float*)d_in[6];
    const float* c2b = (const float*)d_in[7];
    const float* lw  = (const float*)d_in[8];
    const float* lb  = (const float*)d_in[9];
    float* out = (float*)d_out;

    char* ws = (char*)d_ws;
    size_t cur = 0;
    auto take = [&](size_t bytes) -> char* {
        char* p = ws + cur;
        cur += (bytes + 255) & ~(size_t)255;
        return p;
    };
    const size_t MiB = 1024 * 1024;

    bf16*  wbuf = (bf16*)take(64 * MiB);                 // Win/U bf16; reused below
    bf16*  wrbf = (bf16*)take(128 * MiB);                // Wr bf16;  reused below
    bf16*  xbf  = (bf16*)take((size_t)1280 * 4096 * 2);
    float* xU   = (float*)take((size_t)19 * 64 * 8192 * 4);
    float* Hst  = (float*)take((size_t)64 * 20 * 8192 * 4);
    bf16*  hbf0 = (bf16*)take((size_t)64 * 8192 * 2);
    bf16*  hbf1 = (bf16*)take((size_t)64 * 8192 * 2);

    // reuse of wbuf region after the xU GEMM:
    bf16* h5d  = (bf16*)wbuf;                             // 20.0 MiB
    bf16* w1bf = (bf16*)((char*)wbuf + 21 * MiB);         // 28 KiB
    bf16* y1   = (bf16*)((char*)wbuf + 22 * MiB);         // 40.0 MiB
    // reuse of wrbf region after the last recurrence GEMM:
    bf16*  w2bf = (bf16*)wrbf;                            // 768 KiB
    bf16*  y2   = (bf16*)((char*)wrbf + 1 * MiB);         // 15.3 MiB
    bf16*  lwT  = (bf16*)((char*)wrbf + 18 * MiB);        // 3.1 MiB
    float* ylin = (float*)((char*)wrbf + 22 * MiB);       // 1.25 MiB

    auto blks = [](size_t n) { return (unsigned)((n + 255) / 256); };

    // 1) Win -> bf16; x -> bf16 rows; Wr -> bf16
    cvt_f32_bf16_kernel<<<blks((size_t)4096 * 8192), 256, 0, stream>>>(Win, wbuf, (size_t)4096 * 8192);
    prep_x_kernel<<<blks((size_t)1280 * 4096), 256, 0, stream>>>(x, xbf);
    cvt_f32_bf16_kernel<<<blks((size_t)8192 * 8192), 256, 0, stream>>>(Wr, wrbf, (size_t)8192 * 8192);

    // 2) h0 = tanh(x0 @ Win): M=64,N=8192,K=4096 -> H[:,0,:] f32 + hbf0
    gemm_bf16_wmma<<<dim3(8192 / 128, 1), 256, 0, stream>>>(
        xbf, 4096, wbuf, 8192, Hst, 20 * 8192, nullptr, 0, hbf0, 8192, 4096, 1);

    // 3) xU = x[1:] @ U: M=1216,N=8192,K=4096 (overwrites wbuf with U first)
    cvt_f32_bf16_kernel<<<blks((size_t)4096 * 8192), 256, 0, stream>>>(U, wbuf, (size_t)4096 * 8192);
    gemm_bf16_wmma<<<dim3(8192 / 128, 1216 / 64), 256, 0, stream>>>(
        xbf + (size_t)64 * 4096, 4096, wbuf, 8192, xU, 8192, nullptr, 0, nullptr, 0, 4096, 0);

    // 4) recurrence: h_t = tanh(xU[t-1] + h @ Wr), ping-pong bf16 h
    //    (Wr bf16 = 128 MB fits in 192 MB L2 -> 19 re-reads stream from L2)
    for (int t = 1; t < 20; ++t) {
        const bf16* hp = (t & 1) ? hbf0 : hbf1;
        bf16*       hn = (t & 1) ? hbf1 : hbf0;
        gemm_bf16_wmma<<<dim3(8192 / 128, 1), 256, 0, stream>>>(
            hp, 8192, wrbf, 8192,
            Hst + (size_t)t * 8192, 20 * 8192,
            xU + (size_t)(t - 1) * 64 * 8192, 8192,
            hn, 8192, 8192, 1);
    }

    // 5) H -> (B,8,20,32,32) bf16 ; conv1 as implicit GEMM (K 216 -> pad 224)
    relayout_h_kernel<<<blks((size_t)64 * 8 * 20 * 1024), 256, 0, stream>>>(Hst, h5d);
    prep_wconv_kernel<<<blks(64 * 224), 256, 0, stream>>>(c1w, w1bf, 216, 224, 64 * 224);
    conv_igemm_wmma<<<dim3(327680 / 128, 1), 256, 0, stream>>>(
        h5d, w1bf, c1b, y1,
        8, 20, 32, 32, 64, 16, 16, 3, 3, 3, 1, 1, 1, 216, 224);

    // 6) conv2 as implicit GEMM (K=3072): (B,64,20,16,16) -> (B,128,20,7,7)
    prep_wconv_kernel<<<blks(128 * 3072), 256, 0, stream>>>(c2w, w2bf, 3072, 3072, 128 * 3072);
    conv_igemm_wmma<<<dim3(62720 / 128, 128 / 64), 256, 0, stream>>>(
        y1, w2bf, c2b, y2,
        64, 20, 16, 16, 128, 7, 7, 3, 4, 4, 1, 0, 0, 3072, 3072);

    // 7) linear: y2 viewed flat as (1280, 6272) @ lin_w^T (6272,256)
    prep_linT_kernel<<<blks(6272 * 256), 256, 0, stream>>>(lw, lwT);
    gemm_bf16_wmma<<<dim3(256 / 128, 1280 / 64), 256, 0, stream>>>(
        y2, 6272, lwT, 256, ylin, 256, nullptr, 0, nullptr, 0, 6272, 0);

    // 8) + lin_b, split mu / log_var into d_out
    split_out_kernel<<<blks(1280 * 256), 256, 0, stream>>>(ylin, lb, out);
}
